// EGCN_41223096107020
// MI455X (gfx1250) — matrix operand order repaired and checked by reading, not compiled, not run on hardware
//
#include <hip/hip_runtime.h>

#define D 128
#define NLAYERS 5
#define NGRAPHS 8192
#define ATOMV 119
#define BONDV 5
#define MTILE 4                      // 16-row tiles per block => 64 rows/block

typedef __attribute__((ext_vector_type(16))) __bf16 v16bf;
typedef __attribute__((ext_vector_type(8)))  float  v8f;

struct Pack32 { uint4 lo, hi; };   // 32 bytes == v16bf

__device__ inline unsigned short f2bf(float f) {
  unsigned int u = __builtin_bit_cast(unsigned int, f);
  u += 0x7fffu + ((u >> 16) & 1u);           // round-to-nearest-even
  return (unsigned short)(u >> 16);
}

__device__ inline void atomAddF(float* p, float v) {
  __hip_atomic_fetch_add(p, v, __ATOMIC_RELAXED, __HIP_MEMORY_SCOPE_AGENT);
}

__device__ inline v16bf load_frag32(const unsigned short* p) {
  Pack32 pk;
  pk.lo = *(const uint4*)(p);
  pk.hi = *(const uint4*)(p + 8);
  return __builtin_bit_cast(v16bf, pk);
}

// A fragment (16-bit A 16x32): lane r holds row m0+r; e<8 -> K=kb+sel*8+e,
// e>=8 -> K=kb+16+sel*8+(e-8). Both halves are 8 contiguous bf16 = one uint4.
__device__ inline v16bf load_afrag(const unsigned short* rowp, int kb, int sel) {
  Pack32 pk;
  pk.lo = *(const uint4*)(rowp + kb + sel * 8);
  pk.hi = *(const uint4*)(rowp + kb + 16 + sel * 8);
  return __builtin_bit_cast(v16bf, pk);
}

// ---------------------------------------------------------------- utilities
__global__ void fill_kernel(float* p, float v, int n) {
  int i = blockIdx.x * blockDim.x + threadIdx.x;
  if (i < n) p[i] = v;
}

// deg[row[e]] += 1  (deg pre-initialized to 1.0)
__global__ void deg_kernel(const int* __restrict__ row, float* __restrict__ deg, int E) {
  int e = blockIdx.x * blockDim.x + threadIdx.x;
  if (e < E) atomAddF(&deg[row[e]], 1.0f);
}

__global__ void dinv_kernel(const float* __restrict__ deg, float* __restrict__ dinv, int n) {
  int i = blockIdx.x * blockDim.x + threadIdx.x;
  if (i < n) dinv[i] = rsqrtf(deg[i]);
}

__global__ void norm_kernel(const int* __restrict__ row, const int* __restrict__ col,
                            const float* __restrict__ dinv, float* __restrict__ nrm, int E) {
  int e = blockIdx.x * blockDim.x + threadIdx.x;
  if (e < E) nrm[e] = dinv[row[e]] * dinv[col[e]];
}

// Pre-swizzle W (fp32 row-major [K=128][N=128]) into bf16 B-fragment order:
// idx = ((wt*4 + kt)*32 + lane)*16 + e ; n = wt*16 + lane%16 ;
// K = kt*32 + (lane/16)*8 + (e&7) + (e/8)*16   (matches 16-bit B 32x16 wave32 layout)
__global__ void wfrag_kernel(const float* __restrict__ Ws, const float* __restrict__ outW,
                             unsigned short* __restrict__ Wfrag) {
  int idx = blockIdx.x * blockDim.x + threadIdx.x;
  if (idx >= 6 * 16384) return;
  int mat = idx / 16384;
  int t   = idx % 16384;
  int e    = t & 15;
  int lane = (t >> 4) & 31;
  int kt   = (t >> 9) & 3;
  int wt   = t >> 11;
  int n = wt * 16 + (lane & 15);
  int k = kt * 32 + (lane >> 4) * 8 + (e & 7) + (e >> 3) * 16;
  const float* src = (mat < 5) ? (Ws + mat * D * D) : outW;
  Wfrag[idx] = f2bf(src[k * D + n]);
}

// AtomEncoder: h[n,:] = sum_c atom_emb[c][x_feat[n,c]]  -> bf16
__global__ void atom_encode_kernel(const int* __restrict__ xf, const float* __restrict__ aemb,
                                   unsigned short* __restrict__ hb, int N) {
  int idx = blockIdx.x * blockDim.x + threadIdx.x;
  if (idx >= N * 32) return;
  int n = idx >> 5;
  int d = (idx & 31) * 4;
  float sx = 0.f, sy = 0.f, sz = 0.f, sw = 0.f;
#pragma unroll
  for (int c = 0; c < 9; ++c) {
    int id = xf[n * 9 + c];
    float4 v = *(const float4*)(aemb + ((size_t)(c * ATOMV + id)) * D + d);
    sx += v.x; sy += v.y; sz += v.z; sw += v.w;
  }
  unsigned int lo = (unsigned int)f2bf(sx) | ((unsigned int)f2bf(sy) << 16);
  unsigned int hi = (unsigned int)f2bf(sz) | ((unsigned int)f2bf(sw) << 16);
  *(uint2*)(hb + (size_t)n * D + d) = make_uint2(lo, hi);
}

// ------------------------------------------------------------- WMMA GEMM
// C[M,128] = A[M,128](bf16) @ W(bf16, fragment order) + bias ; fp32 out.
// Block = 256 thr = 8 waves; block owns 64 rows (4 x 16-row tiles),
// wave wv owns cols [16wv,16wv+16). B fragments stay in registers and are
// reused across the 4 row tiles (16 WMMAs per wave).
__global__ __launch_bounds__(256) void gemm_bias_kernel(
    const unsigned short* __restrict__ A, const unsigned short* __restrict__ Bfrag,
    const float* __restrict__ bias, float* __restrict__ Cout, int M) {
  int m0   = blockIdx.x * (16 * MTILE);
  int wv   = threadIdx.x >> 5;
  int lane = threadIdx.x & 31;
  int r    = lane & 15;
  int sel  = lane >> 4;

  v16bf bfr[4];
#pragma unroll
  for (int kt = 0; kt < 4; ++kt)
    bfr[kt] = load_frag32(Bfrag + ((size_t)((wv * 4 + kt) * 32 + lane) * 16));

  v8f acc[MTILE];
#pragma unroll
  for (int mt = 0; mt < MTILE; ++mt) acc[mt] = (v8f){};

#pragma unroll
  for (int mt = 0; mt < MTILE; ++mt) {
    const unsigned short* arow = A + (size_t)(m0 + mt * 16 + r) * D;
#pragma unroll
    for (int kt = 0; kt < 4; ++kt) {
      v16bf a = load_afrag(arow, kt * 32, sel);
      acc[mt] = __builtin_amdgcn_wmma_f32_16x16x32_bf16(false, a, false, bfr[kt],
                                                        (short)0, acc[mt], false, false);
    }
  }

  int n = wv * 16 + r;
  float bn = bias[n];
#pragma unroll
  for (int mt = 0; mt < MTILE; ++mt) {
#pragma unroll
    for (int v = 0; v < 8; ++v) {          // C/D: VGPR v -> row base + v + 8*sel, col n
      int m = m0 + mt * 16 + v + 8 * sel;
      Cout[(size_t)m * D + n] = acc[mt][v] + bn;
    }
  }
}

// aggr[n,:] = relu(hW[n,:] + root) / deg[n]   (self term seeds the scatter)
__global__ void init_aggr_kernel(const float* __restrict__ hW, const float* __restrict__ root,
                                 const float* __restrict__ deg, float* __restrict__ aggr, int N) {
  int idx = blockIdx.x * blockDim.x + threadIdx.x;
  if (idx >= N * 32) return;
  int n = idx >> 5;
  int d = (idx & 31) * 4;
  float invd = 1.0f / deg[n];
  float4 h = *(const float4*)(hW + (size_t)n * D + d);
  float4 rt = *(const float4*)(root + d);
  float4 o;
  o.x = fmaxf(h.x + rt.x, 0.f) * invd;
  o.y = fmaxf(h.y + rt.y, 0.f) * invd;
  o.z = fmaxf(h.z + rt.z, 0.f) * invd;
  o.w = fmaxf(h.w + rt.w, 0.f) * invd;
  *(float4*)(aggr + (size_t)n * D + d) = o;
}

// one wave per edge: msg = norm * relu(hW[row] + bond_sum); atomicAdd into aggr[col]
__global__ __launch_bounds__(256) void edge_scatter_kernel(
    const int* __restrict__ row, const int* __restrict__ col, const int* __restrict__ eattr,
    const float* __restrict__ bond, const float* __restrict__ nrm,
    const float* __restrict__ hW, float* __restrict__ aggr, int E) {
  int e = blockIdx.x * 8 + (threadIdx.x >> 5);
  if (e >= E) return;
  int lane = threadIdx.x & 31;
  int d = lane * 4;
  int rs = row[e], cd = col[e];
  int a0 = eattr[e * 3 + 0], a1 = eattr[e * 3 + 1], a2 = eattr[e * 3 + 2];
  float4 e0 = *(const float4*)(bond + (size_t)(0 * BONDV + a0) * D + d);
  float4 e1 = *(const float4*)(bond + (size_t)(1 * BONDV + a1) * D + d);
  float4 e2 = *(const float4*)(bond + (size_t)(2 * BONDV + a2) * D + d);
  float4 h  = *(const float4*)(hW + (size_t)rs * D + d);
  float nm = nrm[e];
  float mx = nm * fmaxf(h.x + e0.x + e1.x + e2.x, 0.f);
  float my = nm * fmaxf(h.y + e0.y + e1.y + e2.y, 0.f);
  float mz = nm * fmaxf(h.z + e0.z + e1.z + e2.z, 0.f);
  float mw = nm * fmaxf(h.w + e0.w + e1.w + e2.w, 0.f);
  float* dst = aggr + (size_t)cd * D + d;
  atomAddF(dst + 0, mx);
  atomAddF(dst + 1, my);
  atomAddF(dst + 2, mz);
  atomAddF(dst + 3, mw);
}

// BN(eval) + ReLU, write next-layer h in bf16
__global__ void bn_relu_kernel(const float* __restrict__ aggr, const float* __restrict__ gamma,
                               const float* __restrict__ beta, unsigned short* __restrict__ hb, int N) {
  int idx = blockIdx.x * blockDim.x + threadIdx.x;
  if (idx >= N * 32) return;
  int n = idx >> 5;
  int d = (idx & 31) * 4;
  const float s = rsqrtf(1.0f + 1e-5f);
  float4 a = *(const float4*)(aggr + (size_t)n * D + d);
  float4 g = *(const float4*)(gamma + d);
  float4 b = *(const float4*)(beta + d);
  float ox = fmaxf(a.x * (g.x * s) + b.x, 0.f);
  float oy = fmaxf(a.y * (g.y * s) + b.y, 0.f);
  float oz = fmaxf(a.z * (g.z * s) + b.z, 0.f);
  float ow = fmaxf(a.w * (g.w * s) + b.w, 0.f);
  unsigned int lo = (unsigned int)f2bf(ox) | ((unsigned int)f2bf(oy) << 16);
  unsigned int hi = (unsigned int)f2bf(oz) | ((unsigned int)f2bf(ow) << 16);
  *(uint2*)(hb + (size_t)n * D + d) = make_uint2(lo, hi);
}

__global__ void pool_kernel(const float* __restrict__ h, const int* __restrict__ batch,
                            float* __restrict__ sums, float* __restrict__ cnt, int N) {
  int idx = blockIdx.x * blockDim.x + threadIdx.x;
  if (idx >= N * 32) return;
  int n = idx >> 5;
  int q = idx & 31;
  int d = q * 4;
  int g = batch[n];
  float4 a = *(const float4*)(h + (size_t)n * D + d);
  float* dst = sums + (size_t)g * D + d;
  atomAddF(dst + 0, a.x);
  atomAddF(dst + 1, a.y);
  atomAddF(dst + 2, a.z);
  atomAddF(dst + 3, a.w);
  if (q == 0) atomAddF(&cnt[g], 1.0f);
}

__global__ void pool_fin_kernel(const float* __restrict__ sums, const float* __restrict__ cnt,
                                unsigned short* __restrict__ pooled) {
  int idx = blockIdx.x * blockDim.x + threadIdx.x;
  if (idx >= NGRAPHS * 32) return;
  int g = idx >> 5;
  int d = (idx & 31) * 4;
  float inv = 1.0f / fmaxf(cnt[g], 1.0f);
  float4 a = *(const float4*)(sums + (size_t)g * D + d);
  unsigned int lo = (unsigned int)f2bf(a.x * inv) | ((unsigned int)f2bf(a.y * inv) << 16);
  unsigned int hi = (unsigned int)f2bf(a.z * inv) | ((unsigned int)f2bf(a.w * inv) << 16);
  *(uint2*)(pooled + (size_t)g * D + d) = make_uint2(lo, hi);
}

extern "C" void kernel_launch(void* const* d_in, const int* in_sizes, int n_in,
                              void* d_out, int out_size, void* d_ws, size_t ws_size,
                              hipStream_t stream) {
  const int*   x_feat    = (const int*)d_in[0];
  const int*   edge_idx  = (const int*)d_in[1];
  const int*   edge_attr = (const int*)d_in[2];
  const int*   batch     = (const int*)d_in[3];
  const float* atom_emb  = (const float*)d_in[4];
  const float* bond_emb  = (const float*)d_in[5];
  const float* Ws        = (const float*)d_in[6];
  const float* bs        = (const float*)d_in[7];
  const float* root_embs = (const float*)d_in[8];
  const float* bn_gamma  = (const float*)d_in[9];
  const float* bn_beta   = (const float*)d_in[10];
  const float* out_W     = (const float*)d_in[11];
  const float* out_b     = (const float*)d_in[12];

  const int N = in_sizes[0] / 9;
  const int E = in_sizes[1] / 2;
  const int* row = edge_idx;
  const int* col = edge_idx + E;

  char* wsb = (char*)d_ws;
  size_t off = 0;
  auto alloc = [&](size_t bytes) -> void* {
    void* p = wsb + off;
    off += (bytes + 255) & ~(size_t)255;
    return p;
  };
  unsigned short* h_bf   = (unsigned short*)alloc((size_t)N * D * 2);
  float*          hW     = (float*)alloc((size_t)N * D * 4);
  float*          aggr   = (float*)alloc((size_t)N * D * 4);
  unsigned short* Wfrag  = (unsigned short*)alloc((size_t)6 * 16384 * 2);
  float*          deg    = (float*)alloc((size_t)N * 4);
  float*          dinv   = (float*)alloc((size_t)N * 4);
  float*          nrm    = (float*)alloc((size_t)E * 4);
  float*          sums   = (float*)alloc((size_t)NGRAPHS * D * 4);
  float*          cnt    = (float*)alloc((size_t)NGRAPHS * 4);
  unsigned short* pooled = (unsigned short*)alloc((size_t)NGRAPHS * D * 2);

  // --- one-time prep (graph-structure terms are layer-invariant) ---
  fill_kernel<<<(N + 255) / 256, 256, 0, stream>>>(deg, 1.0f, N);
  deg_kernel<<<(E + 255) / 256, 256, 0, stream>>>(row, deg, E);
  dinv_kernel<<<(N + 255) / 256, 256, 0, stream>>>(deg, dinv, N);
  norm_kernel<<<(E + 255) / 256, 256, 0, stream>>>(row, col, dinv, nrm, E);
  wfrag_kernel<<<(6 * 16384 + 255) / 256, 256, 0, stream>>>(Ws, out_W, Wfrag);
  atom_encode_kernel<<<(N * 32 + 255) / 256, 256, 0, stream>>>(x_feat, atom_emb, h_bf, N);

  // --- 5 conv layers ---
  for (int l = 0; l < NLAYERS; ++l) {
    gemm_bias_kernel<<<N / (16 * MTILE), 256, 0, stream>>>(
        h_bf, Wfrag + (size_t)l * 16384, bs + l * D, hW, N);
    init_aggr_kernel<<<(N * 32 + 255) / 256, 256, 0, stream>>>(hW, root_embs + l * D,
                                                               deg, aggr, N);
    edge_scatter_kernel<<<(E + 7) / 8, 256, 0, stream>>>(
        row, col, edge_attr, bond_emb + (size_t)l * 3 * BONDV * D, nrm, hW, aggr, E);
    if (l < NLAYERS - 1)
      bn_relu_kernel<<<(N * 32 + 255) / 256, 256, 0, stream>>>(
          aggr, bn_gamma + l * D, bn_beta + l * D, h_bf, N);
  }

  // --- global mean pool + output projection ---
  fill_kernel<<<(NGRAPHS * D + 255) / 256, 256, 0, stream>>>(sums, 0.0f, NGRAPHS * D);
  fill_kernel<<<(NGRAPHS + 255) / 256, 256, 0, stream>>>(cnt, 0.0f, NGRAPHS);
  pool_kernel<<<(N * 32 + 255) / 256, 256, 0, stream>>>(aggr, batch, sums, cnt, N);
  pool_fin_kernel<<<(NGRAPHS * 32 + 255) / 256, 256, 0, stream>>>(sums, cnt, pooled);
  gemm_bias_kernel<<<NGRAPHS / (16 * MTILE), 256, 0, stream>>>(
      pooled, Wfrag + (size_t)5 * 16384, out_b, (float*)d_out, NGRAPHS);
}
